// AdditiveAttention_17540646437189
// MI455X (gfx1250) — matrix operand order repaired and checked by reading, not compiled, not run on hardware
//
#include <hip/hip_runtime.h>
#include <hip/hip_bf16.h>

// Problem sizes (from reference): B=32, S=2048, DH=1024, DF=1024, DM=512
#define BB   32
#define SS   2048
#define DH   1024
#define DF   1024
#define DM   512
#define MTOT (BB * SS)          // 65536 rows of the big GEMM

typedef __attribute__((ext_vector_type(16))) __bf16 v16bf;
typedef __attribute__((ext_vector_type(8)))  __bf16 v8bf;
typedef __attribute__((ext_vector_type(8)))  float  v8f;

static __device__ __forceinline__ unsigned short f2bf_bits(float f) {
    unsigned u = __float_as_uint(f);
    unsigned r = u + 0x7FFFu + ((u >> 16) & 1u);   // round-to-nearest-even
    return (unsigned short)(r >> 16);
}

// ---------------------------------------------------------------- zero fill
__global__ __launch_bounds__(256) void zero4_kernel(float4* __restrict__ p) {
    p[(size_t)blockIdx.x * 256 + threadIdx.x] = make_float4(0.f, 0.f, 0.f, 0.f);
}

// ------------------------------------------------- W1f f32 -> bf16 (in ws)
__global__ __launch_bounds__(256) void cvt_w1f_kernel(const float* __restrict__ w,
                                                      __bf16* __restrict__ o) {
    int i = ((int)blockIdx.x * 256 + (int)threadIdx.x) * 4;   // 512*256*4 == DM*DF
    float4 f = *(const float4*)(w + i);
    unsigned lo = (unsigned)f2bf_bits(f.x) | ((unsigned)f2bf_bits(f.y) << 16);
    unsigned hi = (unsigned)f2bf_bits(f.z) | ((unsigned)f2bf_bits(f.w) << 16);
    *(uint2*)(o + i) = make_uint2(lo, hi);
}

// ----------------------------------------- emb_h = hidden @ W1h^T + b1h
__global__ __launch_bounds__(256) void embh_kernel(const float* __restrict__ hidden,
                                                   const float* __restrict__ W1h,
                                                   const float* __restrict__ b1h,
                                                   float* __restrict__ emb_h) {
    int idx = (int)blockIdx.x * 256 + (int)threadIdx.x;   // 0 .. BB*DM-1
    int b = idx >> 9;          // / DM
    int n = idx & (DM - 1);
    const float4* h = (const float4*)(hidden + (size_t)b * DH);
    const float4* w = (const float4*)(W1h + (size_t)n * DH);
    float acc = 0.f;
#pragma unroll 4
    for (int k = 0; k < DH / 4; ++k) {
        float4 a = h[k], c = w[k];
        acc += a.x * c.x + a.y * c.y + a.z * c.z + a.w * c.w;
    }
    emb_h[idx] = acc + b1h[n];
}

// ---------------------------------------------------------------- main WMMA
// logits[m] = sum_n w2[n] * tanh( emb_h[b,n] + b1f[n] + sum_k feats[m,k]*W1f[n,k] )
// Block: 256 threads = 8 waves, 64 rows of feats staged in LDS as bf16.
// Wave w: m-tile = w>>1 (16 rows), n-half = w&1 (256 of 512 cols), 4 passes
// of 4 accumulators (4 x 16 cols), K loop in steps of 32 (one wmma-K).
#define KSTR 1040   // 1024 + 16 bf16 pad to dodge LDS bank conflicts

__global__ __launch_bounds__(256) void attn_logits_wmma_kernel(
        const float* __restrict__ feats,
        const __bf16* __restrict__ w1f_bf,   // [DM, DF] row-major bf16
        const float* __restrict__ emb_h,     // [BB, DM]
        const float* __restrict__ b1f,       // [DM]
        const float* __restrict__ w2,        // [DM]
        float* __restrict__ logits)          // [MTOT], pre-zeroed
{
    extern __shared__ char smem_raw[];
    __bf16* lds = (__bf16*)smem_raw;         // [64][KSTR]

    const int tid = threadIdx.x;
    const int m0  = (int)blockIdx.x * 64;    // row tile base (64 rows)
    const int b   = m0 >> 11;                // / SS ; tile never crosses batch

    // ---- stage 64 x 1024 feats (f32) -> bf16 LDS, coalesced float4 ----
    for (int i = 0; i < 64; ++i) {
        int linear = (i * 256 + tid) * 4;          // 0 .. 65535
        int row = linear >> 10;
        int col = linear & (DF - 1);
        float4 f = *(const float4*)(feats + (size_t)(m0 + row) * DF + col);
        unsigned lo = (unsigned)f2bf_bits(f.x) | ((unsigned)f2bf_bits(f.y) << 16);
        unsigned hi = (unsigned)f2bf_bits(f.z) | ((unsigned)f2bf_bits(f.w) << 16);
        *(uint2*)(lds + row * KSTR + col) = make_uint2(lo, hi);
    }
    __syncthreads();

    const int lane  = tid & 31;
    const int wave  = tid >> 5;
    const int mt    = wave >> 1;        // 0..3  : which 16-row tile
    const int nhalf = wave & 1;         // 0..1  : which 256-col half
    const int lrow  = lane & 15;
    const int lhi   = lane >> 4;
    const int mloc  = mt * 16 + lrow;   // A row this lane owns
    const int kbaseA = lhi * 16;        // A layout: K 0-15|16-31 halves (matches B)
    const int kbaseB = lhi * 16;        // B layout: K 0-15|16-31 halves (matches A)

    float plog[8];
#pragma unroll
    for (int r = 0; r < 8; ++r) plog[r] = 0.f;

    for (int pass = 0; pass < 4; ++pass) {
        const int nbase = nhalf * 256 + pass * 64;
        v8f acc[4] = {v8f{}, v8f{}, v8f{}, v8f{}};

        for (int k0 = 0; k0 < DF; k0 += 32) {
            // A fragment (16x32 bf16) from LDS: elems 0-15 -> K k0+kbaseA+e
            // (contiguous, same order as B)
            const __bf16* ap = lds + mloc * KSTR + k0 + kbaseA;
            v8bf alo = *(const v8bf*)(ap);
            v8bf ahi = *(const v8bf*)(ap + 8);
            v16bf afrag = __builtin_shufflevector(alo, ahi,
                0, 1, 2, 3, 4, 5, 6, 7, 8, 9, 10, 11, 12, 13, 14, 15);

            // B fragments (32x16 bf16): lane col = lrow, K chunk by lane half;
            // contiguous 32B loads from bf16 W1f rows.
            const __bf16* bp = w1f_bf + (size_t)(nbase + lrow) * DF + k0 + kbaseB;
            v16bf b0 = *(const v16bf*)(bp);
            v16bf b1 = *(const v16bf*)(bp + 16 * DF);
            v16bf b2 = *(const v16bf*)(bp + 32 * DF);
            v16bf b3 = *(const v16bf*)(bp + 48 * DF);

            acc[0] = __builtin_amdgcn_wmma_f32_16x16x32_bf16(false, afrag, false, b0,
                        (short)0, acc[0], false, false);
            acc[1] = __builtin_amdgcn_wmma_f32_16x16x32_bf16(false, afrag, false, b1,
                        (short)0, acc[1], false, false);
            acc[2] = __builtin_amdgcn_wmma_f32_16x16x32_bf16(false, afrag, false, b2,
                        (short)0, acc[2], false, false);
            acc[3] = __builtin_amdgcn_wmma_f32_16x16x32_bf16(false, afrag, false, b3,
                        (short)0, acc[3], false, false);
        }

        // fused epilogue: tanh(acc + emb_h + b1f) * w2, accumulate per-row
#pragma unroll
        for (int j = 0; j < 4; ++j) {
            int ncol = nbase + j * 16 + lrow;
            float eh = emb_h[b * DM + ncol];
            float bi = b1f[ncol];
            float wv = w2[ncol];
#pragma unroll
            for (int r = 0; r < 8; ++r)
                plog[r] += wv * tanhf(acc[j][r] + eh + bi);
        }
    }

    // reduce the 16 N-lanes of each half-wave (C layout: M = r + 8*lhi)
#pragma unroll
    for (int r = 0; r < 8; ++r) {
        float v = plog[r];
        v += __shfl_xor(v, 1, 32);
        v += __shfl_xor(v, 2, 32);
        v += __shfl_xor(v, 4, 32);
        v += __shfl_xor(v, 8, 32);
        plog[r] = v;
    }
    if (lrow == 0) {
        int rowbase = m0 + mt * 16 + lhi * 8;
#pragma unroll
        for (int r = 0; r < 8; ++r)
            atomicAdd(&logits[rowbase + r], plog[r]);   // 2 waves per m-tile
    }
}

// ---------------------------------------------------------------- softmax
__global__ __launch_bounds__(256) void softmax_kernel(const float* __restrict__ logits,
                                                      float* __restrict__ probs) {
    __shared__ float red[256];
    const int b = blockIdx.x, tid = threadIdx.x;
    const float* row = logits + (size_t)b * SS;

    float m = -3.4e38f;
    for (int s = tid; s < SS; s += 256) m = fmaxf(m, row[s]);
    red[tid] = m; __syncthreads();
    for (int off = 128; off > 0; off >>= 1) {
        if (tid < off) red[tid] = fmaxf(red[tid], red[tid + off]);
        __syncthreads();
    }
    m = red[0]; __syncthreads();

    float sum = 0.f;
    for (int s = tid; s < SS; s += 256) sum += __expf(row[s] - m);
    red[tid] = sum; __syncthreads();
    for (int off = 128; off > 0; off >>= 1) {
        if (tid < off) red[tid] += red[tid + off];
        __syncthreads();
    }
    float inv = 1.f / red[0];
    for (int s = tid; s < SS; s += 256)
        probs[(size_t)b * SS + s] = __expf(row[s] - m) * inv;
}

// ------------------------------------------------ context = probs @ feats
// grid = (B, 8 s-chunks); block covers all DF with float4 per thread.
__global__ __launch_bounds__(256) void context_kernel(const float* __restrict__ probs,
                                                      const float* __restrict__ feats,
                                                      float* __restrict__ context) {
    const int b = blockIdx.x, sc = blockIdx.y, tid = threadIdx.x;
    const float* fb = feats + ((size_t)b * SS + (size_t)sc * 256) * DF;
    const float* pb = probs + (size_t)b * SS + sc * 256;
    float ax = 0.f, ay = 0.f, az = 0.f, aw = 0.f;
    for (int s = 0; s < 256; ++s) {
        __builtin_prefetch(fb + (size_t)(s + 8) * DF + tid * 4, 0, 1);
        float p = pb[s];
        float4 f = *(const float4*)(fb + (size_t)s * DF + tid * 4);
        ax += p * f.x; ay += p * f.y; az += p * f.z; aw += p * f.w;
    }
    float* c = context + (size_t)b * DF + tid * 4;
    atomicAdd(c + 0, ax); atomicAdd(c + 1, ay);
    atomicAdd(c + 2, az); atomicAdd(c + 3, aw);
}

// ---------------------------------------------------------------- launcher
extern "C" void kernel_launch(void* const* d_in, const int* in_sizes, int n_in,
                              void* d_out, int out_size, void* d_ws, size_t ws_size,
                              hipStream_t stream) {
    const float* hidden = (const float*)d_in[0];   // [B, DH]
    const float* feats  = (const float*)d_in[1];   // [B, S, DF]
    const float* W1h    = (const float*)d_in[2];   // [DM, DH]
    const float* b1h    = (const float*)d_in[3];   // [DM]
    const float* W1f    = (const float*)d_in[4];   // [DM, DF]
    const float* b1f    = (const float*)d_in[5];   // [DM]
    const float* w2     = (const float*)d_in[6];   // [DM]

    float* context = (float*)d_out;                       // [B, DF]
    float* probs   = (float*)d_out + (size_t)BB * DF;     // [B, S]

    char* ws = (char*)d_ws;
    __bf16* w1f_bf = (__bf16*)ws;                                   // 1 MB
    float*  emb_h  = (float*)(ws + (size_t)DM * DF * 2);            // 64 KB
    float*  logits = (float*)(ws + (size_t)DM * DF * 2 + (size_t)BB * DM * 4); // 256 KB

    // prep: zero context (32768 f) and logits (65536 f); convert W1f
    zero4_kernel<<<(BB * DF) / 1024, 256, 0, stream>>>((float4*)context);
    zero4_kernel<<<MTOT / 1024, 256, 0, stream>>>((float4*)logits);
    cvt_w1f_kernel<<<(DM * DF) / 1024, 256, 0, stream>>>(W1f, w1f_bf);

    // emb_h
    embh_kernel<<<(BB * DM) / 256, 256, 0, stream>>>(hidden, W1h, b1h, emb_h);

    // fused GEMM + tanh + w2-dot  (dynamic LDS: 64 rows x KSTR bf16)
    attn_logits_wmma_kernel<<<MTOT / 64, 256, 64 * KSTR * 2, stream>>>(
        feats, w1f_bf, emb_h, b1f, w2, logits);

    // softmax per batch row
    softmax_kernel<<<BB, 256, 0, stream>>>(logits, probs);

    // context
    context_kernel<<<dim3(BB, SS / 256), 256, 0, stream>>>(probs, feats, context);
}
